// DictionaryLearning_71751723647119
// MI455X (gfx1250) — matrix operand order repaired and checked by reading, not compiled, not run on hardware
//
#include <hip/hip_runtime.h>
#include <hip/hip_bf16.h>
#include <math.h>

// ---------------------------------------------------------------------------
// FISTA dictionary-learning solve, fully fused across iterations.
// M=4, DY=64, DX=128, N=32768.
// GEMMs use V_WMMA_F32_16X16X32_BF16 with bf16x3 split (hi*hi + hi*lo + lo*hi,
// f32 accumulate) for ~f32 accuracy at 1/2.7 the matrix-instruction count of
// the f32 K=4 WMMA path.
// ---------------------------------------------------------------------------

typedef __bf16 bf16_t;
typedef __bf16 v16bf __attribute__((ext_vector_type(16)));
typedef __bf16 v8bf  __attribute__((ext_vector_type(8)));
typedef float  v8f   __attribute__((ext_vector_type(8)));

#define M_MATS   4
#define DY       64
#define DX       128
#define NDATA    32768
#define LAMBD    0.1f

#define OUTF_PITCH 20    // f32 out stripe: 16 cols + 4 pad (floats)
#define OUTB_PITCH 136   // bf16 out, column-major: 128 rows + 8 pad (halves)
#define RESB_PITCH 72    // bf16 resid, column-major: 64 rows + 8 pad (halves)

#define WMMA_BF16(A, B, C) \
    __builtin_amdgcn_wmma_f32_16x16x32_bf16(false, (A), false, (B), (short)0, (C), false, false)

__device__ __forceinline__ v16bf lds_vec16(const bf16_t* p) {
    v8bf a = *(const v8bf*)p;        // 16B, 16B-aligned
    v8bf b = *(const v8bf*)(p + 16);
    return __builtin_shufflevector(a, b, 0,1,2,3,4,5,6,7,8,9,10,11,12,13,14,15);
}

// -------------------- kernel 1: product = operator @ D ---------------------
__global__ __launch_bounds__(256)
void dl_product_kernel(const float* __restrict__ op,   // (4,64,64)
                       const float* __restrict__ Dm,   // (64,128)
                       float* __restrict__ prod)       // (4,64,128)
{
    int idx = blockIdx.x * 256 + threadIdx.x;          // 0 .. 32767
    int m   = idx >> 13;
    int rem = idx & 8191;
    int i   = rem >> 7;
    int x   = rem & 127;
    const float* oprow = op + (m * DY + i) * DY;
    float acc = 0.f;
#pragma unroll 8
    for (int k = 0; k < DY; ++k)
        acc = fmaf(oprow[k], Dm[k * DX + x], acc);
    prod[idx] = acc;
}

// ------------- kernel 2: sumsq[m] = || P_m^T P_m ||_F^2 --------------------
__global__ __launch_bounds__(256)
void dl_gram_kernel(const float* __restrict__ prod, float* __restrict__ sumsq)
{
    __shared__ float red[256];
    int m = blockIdx.x;
    const float* P = prod + m * DY * DX;
    float acc = 0.f;
    for (int p = threadIdx.x; p < DX * DX; p += 256) {
        int i = p >> 7, j = p & 127;
        float g = 0.f;
#pragma unroll 8
        for (int k = 0; k < DY; ++k)
            g = fmaf(P[k * DX + i], P[k * DX + j], g);
        acc = fmaf(g, g, acc);
    }
    red[threadIdx.x] = acc;
    __syncthreads();
    for (int s = 128; s > 0; s >>= 1) {
        if (threadIdx.x < (unsigned)s) red[threadIdx.x] += red[threadIdx.x + s];
        __syncthreads();
    }
    if (threadIdx.x == 0) sumsq[m] = red[0];
}

// ----------------------- main fused FISTA kernel ---------------------------
// grid = 4 * 256; block = 256 threads = 8 waves; each wave owns a 16-column
// stripe of a 128-column tile.  No barriers inside the iteration loop.
__global__ __launch_bounds__(256)
void dl_fista_kernel(const float* __restrict__ y,      // (4,64,32768)
                     const float* __restrict__ prod,   // (4,64,128)
                     const float* __restrict__ sumsq,  // (4)
                     float* __restrict__ out,          // (4,128,32768)
                     const int* __restrict__ max_iter_p)
{
    // A-fragments of P (f<32) and P^T (f>=32), hi/lo terms, pre-swizzled into
    // the 16x32 bf16 WMMA lane layout.  Shared by all waves, built once.
    __shared__ v16bf sAfrag[64][32];                       // 64 KB
    __shared__ float  sOutF[8][DX * OUTF_PITCH];           // 80 KB f32 out
    __shared__ __align__(16) bf16_t sOutH[8][16 * OUTB_PITCH];  // 34 KB
    __shared__ __align__(16) bf16_t sOutL[8][16 * OUTB_PITCH];  // 34 KB
    __shared__ __align__(16) bf16_t sResH[8][16 * RESB_PITCH];  // 18 KB
    __shared__ __align__(16) bf16_t sResL[8][16 * RESB_PITCH];  // 18 KB
    __shared__ float  sY[8][4 * 8 * 32];                   // 32 KB

    const int tid  = threadIdx.x;
    const int wave = tid >> 5;
    const int lane = tid & 31;
    const int hi   = lane >> 4;        // 0: lanes 0-15, 1: lanes 16-31
    const int lc   = lane & 15;        // column within the wave's stripe
    const int m    = blockIdx.x >> 8;
    const int blk  = blockIdx.x & 255;
    const int col0 = blk * 128;
    const int maxIter = max_iter_p[0];

    float mx = fmaxf(fmaxf(sumsq[0], sumsq[1]), fmaxf(sumsq[2], sumsq[3]));
    const float step = 1.0f / sqrtf(mx);
    const float thr  = step * LAMBD;

    // ---- build P / P^T bf16 hi/lo A-fragments (once) ----
    // 16-bit A 16x32 layout: element e of lane -> K = (e&7) + ((e>>3)<<4) + hi*8,
    // row M = 16*t + (lane&15).
    const float* Pm = prod + m * DY * DX;
    for (int f = wave; f < 64; f += 8) {
        v16bf frag;
#pragma unroll
        for (int e = 0; e < 16; ++e) {
            int ke = (e & 7) + ((e >> 3) << 4) + hi * 8;
            float v;
            if (f < 32) {              // GEMM1 A = P (M=dy rows, K=dx)
                int t  = f >> 3;
                int kb = (f >> 1) & 3;
                v = Pm[(16 * t + lc) * DX + kb * 32 + ke];
            } else {                   // GEMM2 A = P^T (M=dx rows, K=dy)
                int g  = f - 32;
                int t  = g >> 2;
                int kb = (g >> 1) & 1;
                v = Pm[(kb * 32 + ke) * DX + 16 * t + lc];
            }
            bf16_t h = (bf16_t)v;
            frag[e] = (f & 1) ? (bf16_t)(v - (float)h) : h;
        }
        sAfrag[f][lane] = frag;
    }

    // ---- load y stripe in accumulator order: slot (t,r) -> row 16t+r+8*hi ----
    const float* Ym = y + (size_t)m * DY * NDATA + col0;
    const int cw = wave * 16 + lc;
#pragma unroll
    for (int t = 0; t < 4; ++t)
#pragma unroll
        for (int r = 0; r < 8; ++r) {
            int row = 16 * t + r + 8 * hi;
            sY[wave][(t * 8 + r) * 32 + lane] = Ym[(size_t)row * NDATA + cw];
        }

    // ---- zero-init out (f32 + bf16 hi/lo) and iterate registers ----
    float*  soF = sOutF[wave];
    bf16_t* soH = sOutH[wave];
    bf16_t* soL = sOutL[wave];
    bf16_t* srH = sResH[wave];
    bf16_t* srL = sResL[wave];
    for (int i = lane; i < DX * OUTF_PITCH; i += 32) soF[i] = 0.f;
    for (int i = lane; i < 16 * OUTB_PITCH; i += 32) { soH[i] = (bf16_t)0.f; soL[i] = (bf16_t)0.f; }
    float iterReg[64];
#pragma unroll
    for (int j = 0; j < 64; ++j) iterReg[j] = 0.f;

    __syncthreads();   // publish sAfrag

    float tk = 1.0f;

    for (int it = 0; it < maxIter; ++it) {
        // ========== GEMM1: resid(64x16) = P(64x128) * out(128x16) ==========
        v8f racc[4];
#pragma unroll
        for (int t = 0; t < 4; ++t) { v8f z = {}; racc[t] = z; }

#pragma unroll
        for (int kb = 0; kb < 4; ++kb) {               // K = 128, 32 per WMMA
            const int boff = lc * OUTB_PITCH + kb * 32 + hi * 8;
            v16bf bh = lds_vec16(&soH[boff]);
            v16bf bl = lds_vec16(&soL[boff]);
#pragma unroll
            for (int t = 0; t < 4; ++t) {
                v16bf aH = sAfrag[(t * 4 + kb) * 2 + 0][lane];
                v16bf aL = sAfrag[(t * 4 + kb) * 2 + 1][lane];
                racc[t] = WMMA_BF16(aH, bh, racc[t]);  // hi*hi
                racc[t] = WMMA_BF16(aH, bl, racc[t]);  // hi*lo
                racc[t] = WMMA_BF16(aL, bh, racc[t]);  // lo*hi
            }
        }

        // resid -= y ; split to bf16 hi/lo, column-major, wave-private LDS
#pragma unroll
        for (int t = 0; t < 4; ++t)
#pragma unroll
            for (int r = 0; r < 8; ++r) {
                float v = racc[t][r] - sY[wave][(t * 8 + r) * 32 + lane];
                int row = 16 * t + r + 8 * hi;
                bf16_t h = (bf16_t)v;
                srH[lc * RESB_PITCH + row] = h;
                srL[lc * RESB_PITCH + row] = (bf16_t)(v - (float)h);
            }

        // ========== GEMM2: grad(128x16) = P^T(128x64) * resid(64x16) =======
        v8f gacc[8];
#pragma unroll
        for (int t = 0; t < 8; ++t) { v8f z = {}; gacc[t] = z; }

#pragma unroll
        for (int kb = 0; kb < 2; ++kb) {               // K = 64
            const int boff = lc * RESB_PITCH + kb * 32 + hi * 8;
            v16bf bh = lds_vec16(&srH[boff]);
            v16bf bl = lds_vec16(&srL[boff]);
#pragma unroll
            for (int t = 0; t < 8; ++t) {
                v16bf aH = sAfrag[32 + (t * 2 + kb) * 2 + 0][lane];
                v16bf aL = sAfrag[32 + (t * 2 + kb) * 2 + 1][lane];
                gacc[t] = WMMA_BF16(aH, bh, gacc[t]);
                gacc[t] = WMMA_BF16(aH, bl, gacc[t]);
                gacc[t] = WMMA_BF16(aL, bh, gacc[t]);
            }
        }

        // ========== elementwise FISTA update (lane-local, f32) =============
        float tnew = 0.5f * (1.0f + sqrtf(fmaf(4.0f * tk, tk, 1.0f)));
        float mom  = (tk - 1.0f) / tnew;
        tk = tnew;
#pragma unroll
        for (int t = 0; t < 8; ++t)
#pragma unroll
            for (int r = 0; r < 8; ++r) {
                int row = 16 * t + r + 8 * hi;
                int idx = row * OUTF_PITCH + lc;
                float o  = soF[idx];
                float z  = fmaf(-step, gacc[t][r], o);     // out - step*grad
                float az = fabsf(z) - thr;
                float ni = az > 0.f ? copysignf(az, z) : 0.f;
                float io = iterReg[t * 8 + r];
                float no = fmaf(mom, ni - io, ni);
                iterReg[t * 8 + r] = ni;
                soF[idx] = no;
                bf16_t h = (bf16_t)no;                     // split for next GEMM1
                soH[lc * OUTB_PITCH + row] = h;
                soL[lc * OUTB_PITCH + row] = (bf16_t)(no - (float)h);
            }
        // all per-iteration state is wave-private -> no __syncthreads
    }

    // ---- write out stripe to global ----
    float* Om = out + (size_t)m * DX * NDATA + col0;
#pragma unroll
    for (int t = 0; t < 8; ++t)
#pragma unroll
        for (int r = 0; r < 8; ++r) {
            int row = 16 * t + r + 8 * hi;
            Om[(size_t)row * NDATA + cw] = soF[row * OUTF_PITCH + lc];
        }
}

// ---------------------------------------------------------------------------
extern "C" void kernel_launch(void* const* d_in, const int* in_sizes, int n_in,
                              void* d_out, int out_size, void* d_ws, size_t ws_size,
                              hipStream_t stream) {
    const float* y  = (const float*)d_in[0];   // (4,64,32768)
    const float* op = (const float*)d_in[1];   // (4,64,64)
    const float* Dm = (const float*)d_in[2];   // (64,128)
    const int* max_iter = (const int*)d_in[3]; // scalar
    float* outp = (float*)d_out;               // (4,128,32768)

    float* prod  = (float*)d_ws;               // 4*64*128 floats
    float* sumsq = prod + M_MATS * DY * DX;    // 4 floats

    dl_product_kernel<<<(M_MATS * DY * DX) / 256, 256, 0, stream>>>(op, Dm, prod);
    dl_gram_kernel<<<M_MATS, 256, 0, stream>>>(prod, sumsq);
    dl_fista_kernel<<<M_MATS * (NDATA / 128), 256, 0, stream>>>(y, prod, sumsq, outp, max_iter);
}
